// GINet_3831110828526
// MI455X (gfx1250) — matrix operand order repaired and checked by reading, not compiled, not run on hardware
//
#include <hip/hip_runtime.h>
#include <hip/hip_bf16.h>
#include <stdint.h>

// ---------------------------------------------------------------------------
// GIN forward for MI455X (gfx1250, wave32, WMMA).
// GEMMs: bf16 A/B via v_wmma_f32_16x16x32_bf16, f32 accumulate.
//   - Weights pre-packed once into WMMA B-fragment order (bf16, global) so the
//     GEMM hot loop is pure global_load_b128 -> v_wmma, no LDS, no barriers.
//   - Wave tile 32x64 (8 WMMAs / K-step), block tile 256x64 (8 waves).
// Scatter/gather message passing + pooling: f32 global atomics.
// BatchNorm (training mode): two-pass with atomic partial sums.
// ---------------------------------------------------------------------------

typedef __attribute__((ext_vector_type(16))) __bf16 v16bf;
typedef __attribute__((ext_vector_type(8)))  float  v8f;
typedef unsigned short u16;

__device__ __forceinline__ u16 f2bf(float f) {
  uint32_t u = __float_as_uint(f);
  u += 0x7FFFu + ((u >> 16) & 1u);   // round-to-nearest-even
  return (u16)(u >> 16);
}

// ---------------------------------------------------------------------------
// Embedding sum kernels (one block per row, one thread per channel, IN_CH=128)
// ---------------------------------------------------------------------------
__global__ __launch_bounds__(128)
void k_embed_atom(const int* __restrict__ idx,
                  const float* __restrict__ t0, const float* __restrict__ t1,
                  const float* __restrict__ t2, const float* __restrict__ t3,
                  const float* __restrict__ t4, const float* __restrict__ t5,
                  const float* __restrict__ t6, const float* __restrict__ t7,
                  const float* __restrict__ t8,
                  u16* __restrict__ out) {
  const int n = blockIdx.x;
  const int c = threadIdx.x;
  const int* ip = idx + (size_t)n * 9;
  float s = t0[(size_t)ip[0] * 128 + c] + t1[(size_t)ip[1] * 128 + c]
          + t2[(size_t)ip[2] * 128 + c] + t3[(size_t)ip[3] * 128 + c]
          + t4[(size_t)ip[4] * 128 + c] + t5[(size_t)ip[5] * 128 + c]
          + t6[(size_t)ip[6] * 128 + c] + t7[(size_t)ip[7] * 128 + c]
          + t8[(size_t)ip[8] * 128 + c];
  out[(size_t)n * 128 + c] = f2bf(s);
}

__global__ __launch_bounds__(128)
void k_embed_bond(const int* __restrict__ idx,
                  const float* __restrict__ t0, const float* __restrict__ t1,
                  const float* __restrict__ t2,
                  u16* __restrict__ out) {
  const int e = blockIdx.x;
  const int c = threadIdx.x;
  const int* ip = idx + (size_t)e * 3;
  float s = t0[(size_t)ip[0] * 128 + c] + t1[(size_t)ip[1] * 128 + c]
          + t2[(size_t)ip[2] * 128 + c];
  out[(size_t)e * 128 + c] = f2bf(s);
}

// ---------------------------------------------------------------------------
// Weight pack: f32 W[K,N] row-major -> bf16 fragments in WMMA B layout.
// Fragment (nChunk = n/16, kBlk = k/32) is 512 u16 (32 lanes x 16 halves):
//   lane = (n&15) + ((k&8)<<1), half j = (k&7) + ((k&16)>>1)  ->  P[frag*512 + lane*16 + j]
//   frag = nChunk * (K/32) + kBlk
// Each lane's 16 halves are 32 contiguous bytes -> 2x b128 loads in the GEMM.
// ---------------------------------------------------------------------------
__global__ __launch_bounds__(256)
void k_pack_b(const float* __restrict__ W, u16* __restrict__ P, int K, int N) {
  const int i = blockIdx.x * 256 + threadIdx.x;
  if (i >= K * N) return;
  const int k = i / N;
  const int n = i % N;
  const int frag = (n >> 4) * (K >> 5) + (k >> 5);
  const int lane = (n & 15) + ((k & 8) << 1);
  const int j = (k & 7) + ((k & 16) >> 1);
  P[(size_t)frag * 512 + lane * 16 + j] = f2bf(W[i]);
}

// ---------------------------------------------------------------------------
// bf16 GEMM: C[M,N] f32 = A[M,K] bf16 @ Bp(packed) + bias
// Block = 256 threads (8 waves). Block tile 256x64, wave tile 32x64, K step 32.
// A layout per documented 16-bit A fragment:
//   lanes 0..15: K {0..7, 16..23}; lanes 16..31: K {8..15, 24..31}
// ---------------------------------------------------------------------------
#define GEMM_BM 256
#define GEMM_BN 64

__global__ __launch_bounds__(256)
void k_gemm_bf16(const u16* __restrict__ A, const u16* __restrict__ Bp,
                 const float* __restrict__ bias, float* __restrict__ C,
                 int M, int N, int K) {
  const int tid  = threadIdx.x;
  const int wave = tid >> 5;
  const int lane = tid & 31;
  const int nTilesN = N / GEMM_BN;
  const int tileM = (blockIdx.x / nTilesN) * GEMM_BM;
  const int tileN = (blockIdx.x % nTilesN) * GEMM_BN;
  const int kFr = K >> 5;            // number of 32-wide K blocks

  const int r0   = tileM + wave * 32 + (lane & 15);
  const int aOff = (lane < 16) ? 0 : 8;      // A fragment interleave
  const u16* a0p = A + (size_t)r0 * K + aOff;
  const u16* a1p = a0p + (size_t)16 * K;
  const u16* bb  = Bp + (size_t)(tileN >> 4) * kFr * 512 + lane * 16;

  v8f acc[2][4] = {};

  for (int kb = 0; kb < kFr; ++kb) {
    union F { v16bf v; uint4 q[2]; } a0, a1, b;
    const u16* ap0 = a0p + kb * 32;
    const u16* ap1 = a1p + kb * 32;
    a0.q[0] = *(const uint4*)(ap0);
    a0.q[1] = *(const uint4*)(ap0 + 16);
    a1.q[0] = *(const uint4*)(ap1);
    a1.q[1] = *(const uint4*)(ap1 + 16);
    if (kb + 1 < kFr) {                       // global_prefetch next A block
      __builtin_prefetch(ap0 + 32, 0, 3);
      __builtin_prefetch(ap1 + 32, 0, 3);
    }
#pragma unroll
    for (int c = 0; c < 4; ++c) {
      const u16* bp = bb + ((size_t)c * kFr + kb) * 512;   // this lane's frag
      b.q[0] = *(const uint4*)(bp);
      b.q[1] = *(const uint4*)(bp + 8);
      acc[0][c] = __builtin_amdgcn_wmma_f32_16x16x32_bf16(
                      false, a0.v, false, b.v, (short)0, acc[0][c], false, false);
      acc[1][c] = __builtin_amdgcn_wmma_f32_16x16x32_bf16(
                      false, a1.v, false, b.v, (short)0, acc[1][c], false, false);
    }
  }

  // store: acc half h, VGPR j -> row wave*32 + h*16 + j + (lane<16?0:8)
  const int rBase = tileM + wave * 32 + ((lane < 16) ? 0 : 8);
  const int cBase = tileN + (lane & 15);
#pragma unroll
  for (int h = 0; h < 2; ++h) {
    const int rb = rBase + h * 16;
#pragma unroll
    for (int c = 0; c < 4; ++c) {
      const int col = cBase + c * 16;
      const float bv = bias ? bias[col] : 0.0f;
#pragma unroll
      for (int j = 0; j < 8; ++j)
        C[(size_t)(rb + j) * N + col] = acc[h][c][j] + bv;
    }
  }
}

// ---------------------------------------------------------------------------
// BatchNorm stats: partial sum / sumsq per channel via f32 atomics.
// ---------------------------------------------------------------------------
__global__ __launch_bounds__(256)
void k_bn_stats(const float* __restrict__ H, float* __restrict__ stats,
                int Mrows, int C, int rowsPerBlock) {
  const int r0 = blockIdx.x * rowsPerBlock;
  int r1 = r0 + rowsPerBlock;
  if (r1 > Mrows) r1 = Mrows;
  for (int c = threadIdx.x; c < C; c += blockDim.x) {
    float s = 0.f, sq = 0.f;
    for (int r = r0; r < r1; ++r) {
      const float v = H[(size_t)r * C + c];
      s += v;
      sq += v * v;
    }
    atomicAdd(&stats[c], s);
    atomicAdd(&stats[C + c], sq);
  }
}

// BN + ReLU, emitting bf16 (next GEMM A operand)
__global__ __launch_bounds__(256)
void k_bn_relu_bf16(const float* __restrict__ H, const float* __restrict__ stats,
                    const float* __restrict__ g, const float* __restrict__ be,
                    u16* __restrict__ outB, int Mrows, int C) {
  const size_t i = (size_t)blockIdx.x * blockDim.x + threadIdx.x;
  if (i >= (size_t)Mrows * C) return;
  const int c = (int)(i % C);
  const float inv = 1.0f / (float)Mrows;
  const float mu  = stats[c] * inv;
  const float var = stats[C + c] * inv - mu * mu;
  float v = (H[i] - mu) * rsqrtf(var + 1e-5f) * g[c] + be[c];
  v = fmaxf(v, 0.0f);
  outB[i] = f2bf(v);
}

// BN + ReLU, emitting both f32 (residual / gather source) and bf16 (GEMM A)
__global__ __launch_bounds__(256)
void k_bn_relu_f32_bf16(const float* __restrict__ H, const float* __restrict__ stats,
                        const float* __restrict__ g, const float* __restrict__ be,
                        float* __restrict__ outF, u16* __restrict__ outB,
                        int Mrows, int C) {
  const size_t i = (size_t)blockIdx.x * blockDim.x + threadIdx.x;
  if (i >= (size_t)Mrows * C) return;
  const int c = (int)(i % C);
  const float inv = 1.0f / (float)Mrows;
  const float mu  = stats[c] * inv;
  const float var = stats[C + c] * inv - mu * mu;
  float v = (H[i] - mu) * rsqrtf(var + 1e-5f) * g[c] + be[c];
  v = fmaxf(v, 0.0f);
  outF[i] = v;
  outB[i] = f2bf(v);
}

// ---------------------------------------------------------------------------
// Message + scatter-add: agg[dst] += relu(x[src] + ea[e])  (one block / edge)
// ---------------------------------------------------------------------------
__global__ __launch_bounds__(256)
void k_msg_agg(const float* __restrict__ x, const float* __restrict__ ea,
               const int* __restrict__ ei, float* __restrict__ agg,
               int E, int C) {
  const int e = blockIdx.x;
  const int c = threadIdx.x;
  const int src = ei[e];
  const int dst = ei[E + e];
  const float m = fmaxf(x[(size_t)src * C + c] + ea[(size_t)e * C + c], 0.0f);
  atomicAdd(&agg[(size_t)dst * C + c], m);
}

// h_in = (1+eps)*x + agg  ->  bf16
__global__ __launch_bounds__(256)
void k_combine(const float* __restrict__ x, const float* __restrict__ agg,
               const float* __restrict__ epsp, u16* __restrict__ out,
               size_t total) {
  const size_t i = (size_t)blockIdx.x * blockDim.x + threadIdx.x;
  if (i >= total) return;
  out[i] = f2bf((1.0f + epsp[0]) * x[i] + agg[i]);
}

// global_add_pool: out[frag_batch[n]] += x[n]
__global__ __launch_bounds__(256)
void k_pool(const float* __restrict__ xf, const int* __restrict__ batch,
            float* __restrict__ out, int Nn, int C) {
  const int n = blockIdx.x;
  const int c = threadIdx.x;
  const int g = batch[n];
  atomicAdd(&out[(size_t)g * C + c], xf[(size_t)n * C + c]);
}

// ---------------------------------------------------------------------------
// Host orchestration
// ---------------------------------------------------------------------------
extern "C" void kernel_launch(void* const* d_in, const int* in_sizes, int n_in,
                              void* d_out, int out_size, void* d_ws, size_t ws_size,
                              hipStream_t stream) {
  (void)n_in; (void)ws_size;
  const int IN_CH = 128, HID = 256, HID2 = 512;
  const int Nn = in_sizes[45] / 9;   // nodes
  const int Ee = in_sizes[46] / 3;   // edges

  // ---- input unpack (insertion-order flattening of setup_inputs dict) ----
  const float* atab[9]; for (int i = 0; i < 9; ++i) atab[i] = (const float*)d_in[i];
  const float* btab[3]; for (int i = 0; i < 3; ++i) btab[i] = (const float*)d_in[9 + i];
  const float* wa = (const float*)d_in[12]; const float* ba = (const float*)d_in[13];
  const float* wb = (const float*)d_in[14]; const float* bb = (const float*)d_in[15];
  const float* wl = (const float*)d_in[43]; const float* bl = (const float*)d_in[44];
  const int* xidx  = (const int*)d_in[45];
  const int* eaidx = (const int*)d_in[46];
  const int* eidx  = (const int*)d_in[47];
  const int* fb    = (const int*)d_in[48];

  // ---- workspace carve ----
  char* p = (char*)d_ws;
  auto carve = [&](size_t bytes) -> char* {
    char* r = p;
    p += (bytes + 255) & ~(size_t)255;
    return r;
  };
  float* x_f32 = (float*)carve((size_t)Nn * HID * 4);    // node features f32
  u16*   xb    = (u16*)  carve((size_t)Nn * HID * 2);    // node features bf16 (GEMM A)
  float* ea    = (float*)carve((size_t)Ee * HID * 4);    // edge features f32
  float* dbuf  = (float*)carve((size_t)Nn * HID * 4);    // agg / h2 / final (reused)
  float* hbuf  = (float*)carve((size_t)Nn * HID2 * 4);   // h1 f32 (also embed staging)
  u16*   hb    = (u16*)  carve((size_t)Nn * HID2 * 2);   // h1 bf16
  float* stats = (float*)carve(4096);                    // 2*512 f32
  // packed bf16 weights (fragment-swizzled)
  u16* waP = (u16*)carve((size_t)IN_CH * HID * 2);
  u16* wbP = (u16*)carve((size_t)IN_CH * HID * 2);
  u16* w1P[3]; u16* w2P[3];
  for (int l = 0; l < 3; ++l) {
    w1P[l] = (u16*)carve((size_t)HID * HID2 * 2);
    w2P[l] = (u16*)carve((size_t)HID2 * HID * 2);
  }
  u16* wlP = (u16*)carve((size_t)HID * HID * 2);

  u16* xe = (u16*)hbuf;                                  // [Nn,128] bf16 staging
  u16* ee = (u16*)((char*)hbuf + ((size_t)64 << 20));    // [Ee,128] bf16 staging

  auto pack = [&](const float* W, u16* P, int K, int Nc) {
    k_pack_b<<<dim3((unsigned)((K * Nc + 255) / 256)), 256, 0, stream>>>(W, P, K, Nc);
  };
  auto gemm = [&](const u16* A, const u16* Bp, const float* bias, float* C,
                  int M, int Nc, int K) {
    dim3 grid((unsigned)((M / GEMM_BM) * (Nc / GEMM_BN)));
    k_gemm_bf16<<<grid, 256, 0, stream>>>(A, Bp, bias, C, M, Nc, K);
  };

  // ---- pack all weights into WMMA B-fragment layout ----
  pack(wa, waP, IN_CH, HID);
  pack(wb, wbP, IN_CH, HID);
  for (int l = 0; l < 3; ++l) {
    pack((const float*)d_in[16 + 9 * l + 1], w1P[l], HID, HID2);
    pack((const float*)d_in[16 + 9 * l + 5], w2P[l], HID2, HID);
  }
  pack(wl, wlP, HID, HID);

  // ---- embeddings + input linears ----
  k_embed_atom<<<dim3(Nn), dim3(IN_CH), 0, stream>>>(
      xidx, atab[0], atab[1], atab[2], atab[3], atab[4], atab[5], atab[6],
      atab[7], atab[8], xe);
  k_embed_bond<<<dim3(Ee), dim3(IN_CH), 0, stream>>>(
      eaidx, btab[0], btab[1], btab[2], ee);
  gemm(xe, waP, ba, x_f32, Nn, HID, IN_CH);
  gemm(ee, wbP, bb, ea,    Ee, HID, IN_CH);

  // ---- GIN layers ----
  const size_t totN  = (size_t)Nn * HID;
  const size_t totN2 = (size_t)Nn * HID2;
  for (int l = 0; l < 3; ++l) {
    const int base = 16 + 9 * l;
    const float* epsp = (const float*)d_in[base + 0];
    const float* b1 = (const float*)d_in[base + 2];
    const float* g1 = (const float*)d_in[base + 3];
    const float* be1 = (const float*)d_in[base + 4];
    const float* b2 = (const float*)d_in[base + 6];
    const float* g2 = (const float*)d_in[base + 7];
    const float* be2 = (const float*)d_in[base + 8];

    hipMemsetAsync(dbuf, 0, totN * 4, stream);
    k_msg_agg<<<dim3(Ee), dim3(HID), 0, stream>>>(x_f32, ea, eidx, dbuf, Ee, HID);
    k_combine<<<dim3((unsigned)(totN / 256)), 256, 0, stream>>>(
        x_f32, dbuf, epsp, xb, totN);

    gemm(xb, w1P[l], b1, hbuf, Nn, HID2, HID);
    hipMemsetAsync(stats, 0, 4096, stream);
    k_bn_stats<<<dim3((Nn + 127) / 128), 256, 0, stream>>>(hbuf, stats, Nn, HID2, 128);
    k_bn_relu_bf16<<<dim3((unsigned)(totN2 / 256)), 256, 0, stream>>>(
        hbuf, stats, g1, be1, hb, Nn, HID2);

    gemm(hb, w2P[l], b2, dbuf, Nn, HID, HID2);
    hipMemsetAsync(stats, 0, 4096, stream);
    k_bn_stats<<<dim3((Nn + 127) / 128), 256, 0, stream>>>(dbuf, stats, Nn, HID, 128);
    k_bn_relu_f32_bf16<<<dim3((unsigned)(totN / 256)), 256, 0, stream>>>(
        dbuf, stats, g2, be2, x_f32, xb, Nn, HID);
  }

  // ---- final linear + global_add_pool ----
  gemm(xb, wlP, bl, dbuf, Nn, HID, HID);
  hipMemsetAsync(d_out, 0, (size_t)out_size * 4, stream);
  k_pool<<<dim3(Nn), dim3(HID), 0, stream>>>(dbuf, fb, (float*)d_out, Nn, HID);
}
